// PAM_73821897883821
// MI455X (gfx1250) — compile-verified, matrix operand order checked by reading, not run on hardware
//
#include <hip/hip_runtime.h>
#include <hip/hip_bf16.h>
#include <stdint.h>

// ---------------------------------------------------------------------------
// Problem constants (B,C,H,W = 2,512,64,64 ; N = H*W = 4096 ; DQK = 64)
// ---------------------------------------------------------------------------
#define BATCH 2
#define CCH   512
#define NSP   4096
#define DQK_  64
#define MTOT  640   // 64 q + 64 k + 512 v rows stacked

typedef __attribute__((ext_vector_type(16))) __bf16 v16bf;
typedef __attribute__((ext_vector_type(8)))  __bf16 v8bf;
typedef __attribute__((ext_vector_type(8)))  float  v8f;
typedef __attribute__((ext_vector_type(4)))  int    v4i;

// ---------------------------------------------------------------------------
// WMMA helper: D = A(16x32 bf16) x B(32x16 bf16) + C(16x16 f32)
// ---------------------------------------------------------------------------
__device__ __forceinline__ v8f wmma_bf16(v16bf a, v16bf b, v8f c) {
    return __builtin_amdgcn_wmma_f32_16x16x32_bf16(
        /*neg_a=*/false, a, /*neg_b=*/false, b,
        /*c_mod=*/(short)0, c, /*reuse_a=*/false, /*reuse_b=*/false);
}

// ---------------------------------------------------------------------------
// Fragment loader. Source: 16 "rows" x 32 "k" tile, k contiguous, row stride
// `ld` (bf16 elements, multiple of 8 -> 16B-aligned loads).
// ISA 7.12.2 (16-bit A 16x32): lane L holds row M = L%16; element e maps to
// K = e + (e>=8 ? 8 : 0) + (L>=16 ? 8 : 0)  -> two contiguous 16B loads.
// Used for A (rows = M) and for B tiles stored column-major (rows = N cols).
// ---------------------------------------------------------------------------
__device__ __forceinline__ v16bf load_frag(const __bf16* __restrict__ p, int ld) {
    const int lane = threadIdx.x & 31;
    const __bf16* base = p + (lane & 15) * ld + ((lane >> 4) << 3);
    union { v16bf v; v8bf h[2]; } u;
    u.h[0] = *(const v8bf*)(base);        // K = h..h+7
    u.h[1] = *(const v8bf*)(base + 16);   // K = 16+h..23+h
    return u.v;
}

// ---------------------------------------------------------------------------
// Async global->LDS copy of one 16B segment per lane (ASYNCcnt-tracked).
// Builtin prototype (from hipcc diagnostic): (v4i AS1*, v4i AS3*, Imm, Imm).
// Cast through integers: generic->AS1 is value-preserving; generic LDS ptr
// carries the LDS byte offset in its low 32 bits (ISA 10.2 aperture calc).
// ---------------------------------------------------------------------------
typedef __attribute__((address_space(1))) v4i* as1_v4i_ptr;
typedef __attribute__((address_space(3))) v4i* as3_v4i_ptr;

__device__ __forceinline__ void async_copy_b128(const __bf16* g, __bf16* l) {
#if __has_builtin(__builtin_amdgcn_global_load_async_to_lds_b128)
    __builtin_amdgcn_global_load_async_to_lds_b128(
        (as1_v4i_ptr)(unsigned long long)(uintptr_t)g,
        (as3_v4i_ptr)(unsigned int)(uintptr_t)l,
        /*offset=*/0, /*cpol=*/0);
#else
    *(v8bf*)l = *(const v8bf*)g;
#endif
}

__device__ __forceinline__ void wait_async_zero() {
#if __has_builtin(__builtin_amdgcn_s_wait_asynccnt)
    __builtin_amdgcn_s_wait_asynccnt(0);
#endif
}

// ---------------------------------------------------------------------------
// fp32 -> bf16 transposing convert: x[b][c][n] -> xT[b][n][c]
// 32x32 LDS tile, coalesced reads and writes.
// ---------------------------------------------------------------------------
__global__ void conv_xt_kernel(const float* __restrict__ x, __bf16* __restrict__ xt) {
    const int b  = blockIdx.z;
    const int c0 = blockIdx.y * 32;
    const int n0 = blockIdx.x * 32;
    __shared__ __bf16 tile[32][33];
    const float* xp  = x  + (size_t)b * CCH * NSP;
    __bf16*      xtp = xt + (size_t)b * NSP * CCH;
    const int t0 = threadIdx.x & 31;
    const int t1 = threadIdx.x >> 5;   // 0..7
#pragma unroll
    for (int k = 0; k < 4; ++k) {
        int c = t1 * 4 + k;
        tile[c][t0] = (__bf16)xp[(size_t)(c0 + c) * NSP + n0 + t0];
    }
    __syncthreads();
#pragma unroll
    for (int k = 0; k < 4; ++k) {
        int n = t1 * 4 + k;
        xtp[(size_t)(n0 + n) * CCH + c0 + t0] = tile[t0][n];
    }
}

__global__ void conv_w_kernel(const float* __restrict__ Wq,
                              const float* __restrict__ Wk,
                              const float* __restrict__ Wv,
                              __bf16* __restrict__ Wb) {
    int i = blockIdx.x * 256 + threadIdx.x;     // over 640*512
    if (i >= MTOT * CCH) return;
    int m = i >> 9, c = i & 511;
    float v = (m < 64) ? Wq[m * CCH + c]
            : (m < 128) ? Wk[(m - 64) * CCH + c]
                        : Wv[(size_t)(m - 128) * CCH + c];
    Wb[i] = (__bf16)v;
}

// ---------------------------------------------------------------------------
// Projection GEMM:  Y[m,n] = sum_c W[m,c] * xT[b,n,c] + bias[m]
// Pure WMMA, no LDS: both operands have K (=c) contiguous in memory.
// block = 128 thr (4 waves); block computes 64m x 64n; wave w -> rows m0+16w.
// Outputs (bf16):
//   m <  64 : q_nd[b][n][m]        (n-major, d contiguous)
//   m < 128 : k_nd[b][n][m-64]     (n-major, d contiguous)
//   else    : v_cn[b][m-128][n]    (c-major, n contiguous)
// ---------------------------------------------------------------------------
__global__ void proj_gemm_kernel(const __bf16* __restrict__ Wb,
                                 const __bf16* __restrict__ xt,
                                 const float*  __restrict__ bq,
                                 const float*  __restrict__ bk,
                                 const float*  __restrict__ bv,
                                 __bf16* __restrict__ qn,
                                 __bf16* __restrict__ kn,
                                 __bf16* __restrict__ vc) {
    const int b    = blockIdx.z;
    const int m0   = blockIdx.y * 64;
    const int n0   = blockIdx.x * 64;
    const int wave = threadIdx.x >> 5;
    const int lane = threadIdx.x & 31;

    const __bf16* xtb = xt + (size_t)b * NSP * CCH;
    const __bf16* wp  = Wb + (size_t)(m0 + wave * 16) * CCH;

    v8f acc[4];
#pragma unroll
    for (int t = 0; t < 4; ++t) acc[t] = v8f{};

    for (int c0 = 0; c0 < CCH; c0 += 32) {
        v16bf a = load_frag(wp + c0, CCH);
#pragma unroll
        for (int t = 0; t < 4; ++t) {
            v16bf bf = load_frag(xtb + (size_t)(n0 + t * 16) * CCH + c0, CCH);
            acc[t] = wmma_bf16(a, bf, acc[t]);
        }
    }

    // epilogue: add bias, scatter to layout-specific buffers
    const int rbase = (lane >> 4) << 3;
#pragma unroll
    for (int t = 0; t < 4; ++t) {
        int n = n0 + t * 16 + (lane & 15);
#pragma unroll
        for (int r = 0; r < 8; ++r) {
            int m = m0 + wave * 16 + rbase + r;
            float bias = (m < 64) ? bq[m] : (m < 128 ? bk[m - 64] : bv[m - 128]);
            __bf16 val = (__bf16)(acc[t][r] + bias);
            if (m < 64)
                qn[(size_t)b * NSP * DQK_ + (size_t)n * DQK_ + m] = val;
            else if (m < 128)
                kn[(size_t)b * NSP * DQK_ + (size_t)n * DQK_ + (m - 64)] = val;
            else
                vc[(size_t)b * CCH * NSP + (size_t)(m - 128) * NSP + n] = val;
        }
    }
}

// ---------------------------------------------------------------------------
// Flash attention:
//   grid = B * (N/16) workgroups, 128 thr (4 waves).
//   Workgroup: 16 query rows, all 512 channels (wave w -> 128 channels).
//   Per 64-col j-block:
//     - each wave kicks off async global->LDS staging of its 128x64 V tile
//     - wave w computes S tile (:,16w..) with 2 WMMAs, shares via LDS
//     - 16 threads run the online-softmax row update (max/sum/scale in LDS)
//     - all waves rescale accumulators, wait asynccnt, then P x V^T (16 WMMA)
// ---------------------------------------------------------------------------
#define S_STRIDE 68   // f32 row stride (pad)
#define P_STRIDE 72   // bf16 row stride: 144B, 16B-aligned, conflict-free
#define V_STRIDE 72   // same padding for the staged V tile

__global__ void attn_kernel(const __bf16* __restrict__ qn,
                            const __bf16* __restrict__ kn,
                            const __bf16* __restrict__ vc,
                            const float*  __restrict__ x,
                            const float*  __restrict__ gamma,
                            float* __restrict__ out) {
    const int bi   = blockIdx.x >> 8;          // 256 i-tiles per batch
    const int i0   = (blockIdx.x & 255) << 4;
    const int wave = threadIdx.x >> 5;
    const int lane = threadIdx.x & 31;

    __shared__ __attribute__((aligned(16))) float  sS[16 * S_STRIDE];
    __shared__ __attribute__((aligned(16))) __bf16 sP[16 * P_STRIDE];
    __shared__ __attribute__((aligned(16))) __bf16 sV[4][128 * V_STRIDE]; // 72KB
    __shared__ float sM[16], sL[16], sScale[16];

    const __bf16* q  = qn + (size_t)bi * NSP * DQK_ + (size_t)i0 * DQK_;
    const __bf16* kt = kn + (size_t)bi * NSP * DQK_;
    const __bf16* v  = vc + (size_t)bi * CCH * NSP + (size_t)(wave * 128) * NSP;
    __bf16* sVw = &sV[wave][0];

    // Q fragments stay resident (16 rows x 64 d -> two k-chunks)
    v16bf qa0 = load_frag(q, DQK_);
    v16bf qa1 = load_frag(q + 32, DQK_);

    v8f acc[8];
#pragma unroll
    for (int t = 0; t < 8; ++t) acc[t] = v8f{};

    if (threadIdx.x < 16) { sM[threadIdx.x] = -3.0e38f; sL[threadIdx.x] = 0.f; }
    __syncthreads();

    for (int j0 = 0; j0 < NSP; j0 += 64) {
        // ---- kick off async staging of this wave's V tile (128c x 64j) ----
        // 1024 16B segments / 32 lanes = 32 async instructions per wave.
        // Safe vs. last iteration's ds reads: those completed before the
        // previous WMMAs consumed them (s_wait_dscnt precedes the wmma).
#pragma unroll
        for (int it = 0; it < 32; ++it) {
            int seg   = it * 32 + lane;
            int c_rel = seg >> 3;
            int js    = (seg & 7) * 8;
            async_copy_b128(v + (size_t)c_rel * NSP + j0 + js,
                            sVw + c_rel * V_STRIDE + js);
        }

        // ---- S tile: columns j0 + 16*wave .. +15 ----
        const __bf16* kp = kt + (size_t)(j0 + wave * 16) * DQK_;
        if (j0 + 64 < NSP) {  // hint next K block into cache
            __builtin_prefetch(kt + (size_t)(j0 + 64 + wave * 16) * DQK_, 0, 1);
        }
        v16bf kb0 = load_frag(kp, DQK_);
        v16bf kb1 = load_frag(kp + 32, DQK_);
        v8f s = v8f{};
        s = wmma_bf16(qa0, kb0, s);
        s = wmma_bf16(qa1, kb1, s);

        {   // scatter S tile to LDS (lane: col = 16w + L%16, rows r + 8*(L>=16))
            int col   = wave * 16 + (lane & 15);
            int rbase = (lane >> 4) << 3;
#pragma unroll
            for (int r = 0; r < 8; ++r) sS[(rbase + r) * S_STRIDE + col] = s[r];
        }
        __syncthreads();

        // ---- online softmax row update (threads 0..15, one row each) ----
        if (threadIdx.x < 16) {
            const int row = threadIdx.x;
            float mOld = sM[row];
            float m = mOld;
#pragma unroll
            for (int j = 0; j < 64; ++j) m = fmaxf(m, sS[row * S_STRIDE + j]);
            float sum = 0.f;
#pragma unroll
            for (int j = 0; j < 64; ++j) {
                float p = __expf(sS[row * S_STRIDE + j] - m);
                sum += p;
                sP[row * P_STRIDE + j] = (__bf16)p;
            }
            float scale = __expf(mOld - m);
            sScale[row] = scale;
            sL[row] = sL[row] * scale + sum;
            sM[row] = m;
        }
        __syncthreads();

        // ---- rescale accumulators by exp(mOld - mNew) per row ----
        {
            int rbase = (lane >> 4) << 3;
            float sc[8];
#pragma unroll
            for (int r = 0; r < 8; ++r) sc[r] = sScale[rbase + r];
#pragma unroll
            for (int t = 0; t < 8; ++t)
#pragma unroll
                for (int r = 0; r < 8; ++r) acc[t][r] *= sc[r];
        }

        // ---- O += P(16x64) x V^T(64 x 128ch of this wave), V from LDS ----
        v16bf pa0 = load_frag(sP, P_STRIDE);
        v16bf pa1 = load_frag(sP + 32, P_STRIDE);
        wait_async_zero();   // V tile staged
#pragma unroll
        for (int t = 0; t < 8; ++t) {
            const __bf16* vp = sVw + (t * 16) * V_STRIDE;
            v16bf vb0 = load_frag(vp, V_STRIDE);
            v16bf vb1 = load_frag(vp + 32, V_STRIDE);
            acc[t] = wmma_bf16(pa0, vb0, acc[t]);
            acc[t] = wmma_bf16(pa1, vb1, acc[t]);
        }
        __syncthreads();   // protect sS/sP for next block
    }

    // ---- epilogue: out[b,c,i] = acc/l * gamma + x ----
    const float g = gamma[0];
    const int rbase = (lane >> 4) << 3;
    float linv[8];
#pragma unroll
    for (int r = 0; r < 8; ++r) linv[r] = 1.0f / sL[rbase + r];
#pragma unroll
    for (int t = 0; t < 8; ++t) {
        int c = wave * 128 + t * 16 + (lane & 15);
        size_t base = (size_t)bi * CCH * NSP + (size_t)c * NSP + i0;
#pragma unroll
        for (int r = 0; r < 8; ++r) {
            int i = rbase + r;
            out[base + i] = acc[t][r] * linv[r] * g + x[base + i];
        }
    }
}

// ---------------------------------------------------------------------------
// Host launcher
// ---------------------------------------------------------------------------
extern "C" void kernel_launch(void* const* d_in, const int* in_sizes, int n_in,
                              void* d_out, int out_size, void* d_ws, size_t ws_size,
                              hipStream_t stream) {
    const float* x     = (const float*)d_in[0];
    const float* Wq    = (const float*)d_in[1];
    const float* bq    = (const float*)d_in[2];
    const float* Wk    = (const float*)d_in[3];
    const float* bk    = (const float*)d_in[4];
    const float* Wv    = (const float*)d_in[5];
    const float* bv    = (const float*)d_in[6];
    const float* gamma = (const float*)d_in[7];
    float* out = (float*)d_out;

    // workspace carve-up (bf16 buffers), 256B aligned
    char* ws = (char*)d_ws;
    size_t off = 0;
    auto take = [&](size_t bytes) { void* p = ws + off;
        off = (off + bytes + 255) & ~(size_t)255; return p; };
    __bf16* xt = (__bf16*)take((size_t)BATCH * NSP * CCH * 2);   // 8 MB, x^T
    __bf16* Wb = (__bf16*)take((size_t)MTOT * CCH * 2);          // 640 KB
    __bf16* qn = (__bf16*)take((size_t)BATCH * NSP * DQK_ * 2);  // 1 MB
    __bf16* kn = (__bf16*)take((size_t)BATCH * NSP * DQK_ * 2);  // 1 MB
    __bf16* vc = (__bf16*)take((size_t)BATCH * CCH * NSP * 2);   // 8 MB
    (void)ws_size; (void)in_sizes; (void)n_in; (void)out_size;

    // 1) fp32 -> bf16 (x transposed to [b][n][c] so GEMM K is contiguous)
    {
        dim3 grid(NSP / 32, CCH / 32, BATCH);
        conv_xt_kernel<<<grid, 256, 0, stream>>>(x, xt);
        int wtotal = MTOT * CCH;
        conv_w_kernel<<<(wtotal + 255) / 256, 256, 0, stream>>>(Wq, Wk, Wv, Wb);
    }
    // 2) fused Q/K/V projection GEMM (WMMA, LDS-free)
    {
        dim3 grid(NSP / 64, MTOT / 64, BATCH);
        proj_gemm_kernel<<<grid, 128, 0, stream>>>(Wb, xt, bq, bk, bv, qn, kn, vc);
    }
    // 3) flash attention + gamma residual (WMMA + async LDS staging)
    {
        dim3 grid(BATCH * (NSP / 16));
        attn_kernel<<<grid, 128, 0, stream>>>(qn, kn, vc, x, gamma, out);
    }
}
